// LIS_model_78752520339645
// MI455X (gfx1250) — compile-verified, matrix operand order checked by reading, not run on hardware
//
#include <hip/hip_runtime.h>
#include <hip/hip_bf16.h>

typedef __attribute__((ext_vector_type(16))) _Float16 v16h;
typedef __attribute__((ext_vector_type(8)))  _Float16 v8h;
typedef __attribute__((ext_vector_type(8)))  float    v8f;

#define DECAY   0.2f
#define INV_TAU (1.0f/80.0f)
#define VTH     0.2f
#define TW      30

union FragH { v16h f; v8h p[2]; };

// ---------------- init / repack ----------------

__global__ void k_zero(uint4* __restrict__ p, long n16) {
  long i = (long)blockIdx.x * blockDim.x + threadIdx.x;
  if (i < n16) {
    uint4 z; z.x = 0u; z.y = 0u; z.z = 0u; z.w = 0u;
    p[i] = z;
  }
}

// conv2_w [32][32][3][3] (oc, ic, ky, kx) -> fp16 [oc][9][ic]  (K = tap-major, ic-contiguous)
__global__ void k_pack_w2(const float* __restrict__ w, _Float16* __restrict__ out) {
  int i = blockIdx.x * blockDim.x + threadIdx.x;
  if (i >= 32 * 9 * 32) return;
  int oc = i / 288; int rp = i % 288; int p = rp / 32; int ic = rp % 32;
  out[i] = (_Float16)w[(oc * 32 + ic) * 9 + p];
}

__global__ void k_pack_fc1(const float* __restrict__ w, _Float16* __restrict__ out, int n) {
  int i = blockIdx.x * blockDim.x + threadIdx.x;
  if (i < n) out[i] = (_Float16)w[i];
}

// fc2_w [10][128] -> fp16 [16][128] zero-padded rows
__global__ void k_pack_fc2(const float* __restrict__ w, _Float16* __restrict__ out) {
  int i = blockIdx.x * blockDim.x + threadIdx.x;
  if (i >= 16 * 128) return;
  int row = i / 128; int k = i % 128;
  out[i] = (row < 10) ? (_Float16)w[row * 128 + k] : (_Float16)0;
}

// ---------------- layer 1: conv1 (recomputed, x is time-invariant) + IF ----------------
// one thread per (b,c,y,x) NCHW element of [512,32,28,28]; 50176 blocks x 256, exact.

__global__ void k_layer1(const float* __restrict__ x, const float* __restrict__ w1,
                         float* __restrict__ c1m, float* __restrict__ v1,
                         _Float16* __restrict__ c1s) {
  int idx = blockIdx.x * blockDim.x + threadIdx.x;
  int xw = idx % 28; int t = idx / 28;
  int yh = t % 28;   t /= 28;
  int c  = t % 32;   int b = t / 32;
  const float* xb = x + b * 784;
  const float* wc = w1 + c * 9;
  float conv = 0.f;
#pragma unroll
  for (int ky = 0; ky < 3; ++ky) {
    int yy = yh + ky - 1;
    if ((unsigned)yy >= 28u) continue;
#pragma unroll
    for (int kx = 0; kx < 3; ++kx) {
      int xx = xw + kx - 1;
      if ((unsigned)xx >= 28u) continue;
      conv += wc[ky * 3 + kx] * xb[yy * 28 + xx];
    }
  }
  float m = c1m[idx] * DECAY * (1.f - (float)c1s[idx]) + conv;
  float v = v1[idx] + m;                 // IF node
  float s = (v >= VTH) ? 1.f : 0.f;
  v1[idx]  = v - s * VTH;
  c1m[idx] = m;
  c1s[idx] = (_Float16)s;
}

// pool 2x2 of c1 spikes (NCHW) -> p1p fp16 zero-halo NHWC [b][16][16][32]
// (interior written each step; halo zeroed by k_zero -> conv2 loads are unconditional)
__global__ void k_pool1(const _Float16* __restrict__ c1s, _Float16* __restrict__ p1p) {
  int idx = blockIdx.x * blockDim.x + threadIdx.x;  // 3,211,264 = 12544*256 exact
  int px = idx % 14; int t = idx / 14;
  int py = t % 14;   t /= 14;
  int c  = t % 32;   int b = t / 32;
  const _Float16* base = c1s + ((b * 32 + c) * 28 + py * 2) * 28 + px * 2;
  float s = (float)base[0] + (float)base[1] + (float)base[28] + (float)base[29];
  p1p[((b * 16 + py + 1) * 16 + (px + 1)) * 32 + c] = (_Float16)(s * 0.25f);
}

// ---------------- conv2 as implicit GEMM via WMMA + IF ----------------
// C[M=100352 pixels, N=32 oc], K=288 = 9 taps x 32 ic. One wave per 16x16 tile.
// All 9 tap addresses are base + compile-time immediate offsets (zero-halo input).

__global__ void k_conv2(const _Float16* __restrict__ p1p, const _Float16* __restrict__ w2h,
                        float* __restrict__ c2m, float* __restrict__ v2,
                        _Float16* __restrict__ c2s) {
  int wave = blockIdx.x * 8 + (threadIdx.x >> 5);   // 1568 blocks * 8 waves = 12544 tiles
  int lane = threadIdx.x & 31;
  int r = lane & 15;
  int hi = lane >> 4;
  int icoff = hi * 8;                                // K-half offset per documented striping
  int tileN = wave & 1;
  int tileM = wave >> 1;                             // 0..6271
  int pixel = tileM * 16 + r;
  int b = pixel / 196; int rem = pixel - b * 196;
  int y = rem / 14;    int xc = rem - y * 14;
  int oc = tileN * 16 + r;

  // center-tap base pointers; tap deltas fold into load immediate offsets
  const _Float16* abase = p1p + ((b * 16 + y + 1) * 16 + (xc + 1)) * 32 + icoff;
  const _Float16* bbase = w2h + oc * 288 + icoff;

  v8f acc;
#pragma unroll
  for (int i = 0; i < 8; ++i) acc[i] = 0.f;

#pragma unroll
  for (int p = 0; p < 9; ++p) {
    const int dy = p / 3 - 1, dx = p % 3 - 1;
    FragH A, B;
    const v8h* ap = (const v8h*)(abase + (dy * 16 + dx) * 32);
    A.p[0] = ap[0];      // K = icoff .. icoff+7
    A.p[1] = ap[2];      // K = icoff+16 .. icoff+23
    const v8h* bp = (const v8h*)(bbase + p * 32);
    B.p[0] = bp[0]; B.p[1] = bp[2];
    acc = __builtin_amdgcn_wmma_f32_16x16x32_f16(false, A.f, false, B.f,
                                                 (short)0, acc, false, false);
  }

#pragma unroll
  for (int i = 0; i < 8; ++i) {
    int pix = tileM * 16 + i + hi * 8;               // C layout: VGPR i -> M = i (+8 for hi lanes)
    int idx = pix * 32 + oc;
    float m = c2m[idx] * DECAY * (1.f - (float)c2s[idx]) + acc[i];
    float v = v2[idx] + m;                           // IF node
    float s = (v >= VTH) ? 1.f : 0.f;
    v2[idx]  = v - s * VTH;
    c2m[idx] = m;
    c2s[idx] = (_Float16)s;
  }
}

// pool 2x2 of c2 spikes ([pixel][oc]) -> f fp16 [512][1568] in NCHW-flatten order (c*49+y*7+x)
__global__ void k_pool2(const _Float16* __restrict__ c2s, _Float16* __restrict__ fh) {
  int idx = blockIdx.x * blockDim.x + threadIdx.x;   // 802,816 = 3136*256 exact
  int fi = idx % 1568; int b = idx / 1568;
  int c = fi / 49; int rr = fi % 49; int py = rr / 7; int px = rr % 7;
  int p00 = (b * 14 + py * 2) * 14 + px * 2;
  float s = (float)c2s[p00 * 32 + c] + (float)c2s[(p00 + 1) * 32 + c] +
            (float)c2s[(p00 + 14) * 32 + c] + (float)c2s[(p00 + 15) * 32 + c];
  fh[idx] = (_Float16)(s * 0.25f);
}

// ---------------- fc1 GEMM via WMMA + LIF ----------------
// C[512,128] = f[512,1568] x fc1_w^T ; 32 M-tiles x 8 N-tiles = 256 waves, 49 K-chunks.

__global__ void k_fc1(const _Float16* __restrict__ fh, const _Float16* __restrict__ w,
                      float* __restrict__ h1m, float* __restrict__ v3,
                      _Float16* __restrict__ h1s) {
  int wave = blockIdx.x * 8 + (threadIdx.x >> 5);    // 32 blocks * 8 waves
  int lane = threadIdx.x & 31;
  int r = lane & 15; int hi = lane >> 4; int koff = hi * 8;
  int tileN = wave & 7; int tileM = wave >> 3;
  int row = tileM * 16 + r;
  int col = tileN * 16 + r;

  const _Float16* abase = fh + row * 1568 + koff;
  const _Float16* bbase = w + col * 1568 + koff;

  v8f acc;
#pragma unroll
  for (int i = 0; i < 8; ++i) acc[i] = 0.f;

  for (int kc = 0; kc < 49; ++kc) {
    FragH A, B;
    const v8h* ap = (const v8h*)(abase + kc * 32);
    A.p[0] = ap[0]; A.p[1] = ap[2];
    const v8h* bp = (const v8h*)(bbase + kc * 32);
    B.p[0] = bp[0]; B.p[1] = bp[2];
    acc = __builtin_amdgcn_wmma_f32_16x16x32_f16(false, A.f, false, B.f,
                                                 (short)0, acc, false, false);
  }

#pragma unroll
  for (int i = 0; i < 8; ++i) {
    int ridx = tileM * 16 + i + hi * 8;
    int idx = ridx * 128 + col;
    float m = h1m[idx] * DECAY * (1.f - (float)h1s[idx]) + acc[i];
    float v = v3[idx] + (m - v3[idx]) * INV_TAU;     // LIF node
    float s = (v >= VTH) ? 1.f : 0.f;
    v3[idx]  = v - s * VTH;
    h1m[idx] = m;
    h1s[idx] = (_Float16)s;                          // spike is exact in fp16
  }
}

// ---------------- fc2 GEMM via WMMA (N padded to 16) + LIF + h2sum ----------------

__global__ void k_fc2(const _Float16* __restrict__ h1s, const _Float16* __restrict__ w,
                      float* __restrict__ h2m, float* __restrict__ v4,
                      float* __restrict__ h2s, float* __restrict__ h2sum) {
  int wave = blockIdx.x * 8 + (threadIdx.x >> 5);    // 4 blocks * 8 waves = 32 M-tiles
  int lane = threadIdx.x & 31;
  int r = lane & 15; int hi = lane >> 4; int koff = hi * 8;
  int row = wave * 16 + r;

  const _Float16* abase = h1s + row * 128 + koff;
  const _Float16* bbase = w + r * 128 + koff;

  v8f acc;
#pragma unroll
  for (int i = 0; i < 8; ++i) acc[i] = 0.f;

#pragma unroll
  for (int kc = 0; kc < 4; ++kc) {
    FragH A, B;
    const v8h* ap = (const v8h*)(abase + kc * 32);
    A.p[0] = ap[0]; A.p[1] = ap[2];
    const v8h* bp = (const v8h*)(bbase + kc * 32);
    B.p[0] = bp[0]; B.p[1] = bp[2];
    acc = __builtin_amdgcn_wmma_f32_16x16x32_f16(false, A.f, false, B.f,
                                                 (short)0, acc, false, false);
  }

  if (r < 10) {
#pragma unroll
    for (int i = 0; i < 8; ++i) {
      int ridx = wave * 16 + i + hi * 8;
      int idx = ridx * 10 + r;
      float m = h2m[idx] * DECAY * (1.f - h2s[idx]) + acc[i];
      float v = v4[idx] + (m - v4[idx]) * INV_TAU;   // LIF node
      float s = (v >= VTH) ? 1.f : 0.f;
      v4[idx]    = v - s * VTH;
      h2m[idx]   = m;
      h2s[idx]   = s;
      h2sum[idx] += s;
    }
  }
}

// ---------------- finalize: d_out = concat(h2sum/30, c1s_last/30) ----------------

__global__ void k_finalize(const float* __restrict__ h2sum, const _Float16* __restrict__ c1s,
                           float* __restrict__ out, int n) {
  int i = blockIdx.x * blockDim.x + threadIdx.x;
  if (i >= n) return;
  if (i < 5120) out[i] = h2sum[i] * (1.0f / TW);
  else          out[i] = (float)c1s[i - 5120] * (1.0f / TW);
}

// ---------------- launcher ----------------

extern "C" void kernel_launch(void* const* d_in, const int* in_sizes, int n_in,
                              void* d_out, int out_size, void* d_ws, size_t ws_size,
                              hipStream_t stream) {
  (void)in_sizes; (void)n_in; (void)out_size; (void)ws_size;

  const float* x    = (const float*)d_in[0];   // [512,1,28,28]
  const float* w1   = (const float*)d_in[1];   // [32,1,3,3]
  const float* w2   = (const float*)d_in[2];   // [32,32,3,3]
  const float* fc1w = (const float*)d_in[3];   // [128,1568]
  const float* fc2w = (const float*)d_in[4];   // [10,128]
  // d_in[5] = time_window (device scalar); fixed to TW=30 (graph capture bakes trip count).

  char* ws = (char*)d_ws;
  size_t off = 0;
  auto alloc = [&](size_t bytes) -> char* {
    char* p = ws + off;
    off += (bytes + 255) & ~(size_t)255;
    return p;
  };

  float*    c1m  = (float*)   alloc(12845056ull * 4);
  float*    v1   = (float*)   alloc(12845056ull * 4);
  _Float16* c1s  = (_Float16*)alloc(12845056ull * 2);
  float*    c2m  = (float*)   alloc(3211264ull * 4);
  float*    v2   = (float*)   alloc(3211264ull * 4);
  _Float16* c2s  = (_Float16*)alloc(3211264ull * 2);
  _Float16* p1p  = (_Float16*)alloc(4194304ull * 2);   // [512][16][16][32] zero-halo
  _Float16* fh   = (_Float16*)alloc(802816ull * 2);
  float*    h1m  = (float*)   alloc(65536ull * 4);
  float*    v3   = (float*)   alloc(65536ull * 4);
  _Float16* h1s  = (_Float16*)alloc(65536ull * 2);
  float*    h2m  = (float*)   alloc(5120ull * 4);
  float*    v4   = (float*)   alloc(5120ull * 4);
  float*    h2s  = (float*)   alloc(5120ull * 4);
  float*    h2sum= (float*)   alloc(5120ull * 4);
  size_t state_bytes = off;                       // everything above is zero-initialized
  _Float16* w2h   = (_Float16*)alloc(9216ull * 2);
  _Float16* fc1wh = (_Float16*)alloc(200704ull * 2);
  _Float16* fc2wh = (_Float16*)alloc(2048ull * 2);

  long n16 = (long)(state_bytes / 16);
  k_zero<<<(int)((n16 + 255) / 256), 256, 0, stream>>>((uint4*)ws, n16);
  k_pack_w2 <<<(9216 + 255) / 256, 256, 0, stream>>>(w2, w2h);
  k_pack_fc1<<<(200704 + 255) / 256, 256, 0, stream>>>(fc1w, fc1wh, 200704);
  k_pack_fc2<<<(2048 + 255) / 256, 256, 0, stream>>>(fc2w, fc2wh);

  for (int t = 0; t < TW; ++t) {
    k_layer1<<<50176, 256, 0, stream>>>(x, w1, c1m, v1, c1s);
    k_pool1 <<<12544, 256, 0, stream>>>(c1s, p1p);
    k_conv2 <<<1568,  256, 0, stream>>>(p1p, w2h, c2m, v2, c2s);
    k_pool2 <<<3136,  256, 0, stream>>>(c2s, fh);
    k_fc1   <<<32,    256, 0, stream>>>(fh, fc1wh, h1m, v3, h1s);
    k_fc2   <<<4,     256, 0, stream>>>(h1s, fc2wh, h2m, v4, h2s, h2sum);
  }

  k_finalize<<<(12850176 + 255) / 256, 256, 0, stream>>>(h2sum, c1s, (float*)d_out, 12850176);
}
